// ModelPart1_90305982366371
// MI455X (gfx1250) — matrix operand order repaired and checked by reading, not compiled, not run on hardware
//
#include <hip/hip_runtime.h>
#include <hip/hip_bf16.h>

// ---------------------------------------------------------------------------
// CDNA5 / gfx1250 WMMA types
// ---------------------------------------------------------------------------
typedef __attribute__((ext_vector_type(16))) __bf16        v16bf;
typedef __attribute__((ext_vector_type(8)))  float         v8f;
typedef __attribute__((ext_vector_type(8)))  unsigned int  v8u;
typedef __attribute__((ext_vector_type(4)))  unsigned int  v4u;

#define BN_EPS_F     0.001f
#define BBOX_CLIP_F  4.135166556742356f
#define IMG_WH       512
#define NMS_TH       0.7f
#define MIN_SZ       16.0f
#define PRE_NMS_N    2000
#define POST_NMS_N   300
#define NPROP        36864      // 64*64*9
#define SORT_N       65536

// float -> bf16 (round to nearest even)
__device__ __forceinline__ unsigned short f2bf(float f) {
    unsigned int u = __float_as_uint(f);
    u += 0x7FFFu + ((u >> 16) & 1u);
    return (unsigned short)(u >> 16);
}

// ---------------------------------------------------------------------------
// elementwise f32 -> bf16
// ---------------------------------------------------------------------------
__global__ void cvt_bf16_kernel(const float* __restrict__ in,
                                unsigned short* __restrict__ out, int n) {
    int i = blockIdx.x * blockDim.x + threadIdx.x;
    if (i < n) out[i] = f2bf(in[i]);
}

// ---------------------------------------------------------------------------
// pack HWIO fp32 weights into per-wave B fragments:
// out[kt][nt][lane][16] bf16  (lane holds column N = nt*16 + (lane&15),
// half-wave h holds K 8h..8h+7 then 16+8h..23+8h of the 32-K tile).  The [kt][nt] order makes
// all sub-tile fragments of one K-step contiguous for the NTW-wide conv loop.
// ---------------------------------------------------------------------------
__global__ void pack_w_kernel(const float* __restrict__ w,
                              unsigned short* __restrict__ out,
                              int Ktot, int Cout, int Ntiles) {
    int lane = threadIdx.x & 31;
    int kt   = blockIdx.x;
    int nt   = blockIdx.y;
    int half = lane >> 4;
    int n    = nt * 16 + (lane & 15);
    unsigned short* dst = out + (((size_t)kt * Ntiles + nt) * 32 + lane) * 16;
#pragma unroll
    for (int e = 0; e < 16; ++e) {
        int   k = kt * 32 + (e < 8 ? half * 8 + e : 16 + half * 8 + (e - 8));
        float f = (n < Cout && k < Ktot) ? w[(size_t)k * Cout + n] : 0.0f;
        dst[e] = f2bf(f);
    }
}

// ---------------------------------------------------------------------------
// Implicit-GEMM conv (SAME padding, square SxS image, NHWC) with WMMA bf16.
// One wave computes a 16(pixel) x (16*NTW)(channel) output tile: the A
// fragment (two b128 loads) is loaded once per 32-K step and reused across
// NTW WMMAs.  K loop is ky/kx/ct nested (no integer division in the loop).
// Epilogue: + bias, optional ReLU, optional BN; writes f32 and/or bf16 with a
// channel offset/stride (used for the inception concat).
// ---------------------------------------------------------------------------
template <int NTW>
__global__ void conv_wmma_kernel(
    const unsigned short* __restrict__ act, int Slog2, int Cin,
    const unsigned short* __restrict__ wpack, int Ktiles, int Ntiles,
    int KH, int KW,
    const float* __restrict__ bias,
    const float* __restrict__ bn_g, const float* __restrict__ bn_b,
    const float* __restrict__ bn_m, const float* __restrict__ bn_v,
    int do_relu,
    float* __restrict__ out_f32, unsigned short* __restrict__ out_bf16,
    int Cout, int outStride, int chanOff) {

    const int lane  = threadIdx.x & 31;
    const int mtile = blockIdx.x;
    const int ntg   = blockIdx.y;
    const int half  = lane >> 4;
    const int l15   = lane & 15;
    const int S     = 1 << Slog2;

    const int pix = mtile * 16 + l15;          // A-matrix row (output pixel)
    const int py  = pix >> Slog2, px = pix & (S - 1);
    const int pad_h = KH >> 1, pad_w = KW >> 1;
    const int ctn   = Cin >> 5;                // 32-K tiles per kernel position

    v8f acc[NTW];
    int toff[NTW];                             // per-subtile byte-ish offsets
#pragma unroll
    for (int t = 0; t < NTW; ++t) {
        acc[t] = v8f{};
        int nt = ntg * NTW + t;
        if (nt > Ntiles - 1) nt = Ntiles - 1;  // clamp loads; stores guarded
        toff[t] = nt * 512 + lane * 16;        // in ushort units
    }
    const unsigned short* wbase = wpack;
    const size_t ktStride = (size_t)Ntiles * 512;   // ushorts per K-step

    for (int ky = 0; ky < KH; ++ky) {
        for (int kx = 0; kx < KW; ++kx) {
            int  iy  = py + ky - pad_h, ix = px + kx - pad_w;
            bool inb = ((unsigned)iy < (unsigned)S) & ((unsigned)ix < (unsigned)S);
            const unsigned short* ap =
                act + ((size_t)((iy << Slog2) + ix) * Cin + half * 8);
            for (int ct = 0; ct < ctn; ++ct) {
                // A fragment: two contiguous 16B runs per lane
                v4u lo = {}, hi = {};
                if (inb) {
                    const unsigned short* a0 = ap + (ct << 5);
                    lo = *(const v4u*)a0;
                    hi = *(const v4u*)(a0 + 16);
                }
                v8u au = __builtin_shufflevector(lo, hi, 0, 1, 2, 3, 4, 5, 6, 7);
                v16bf a = __builtin_bit_cast(v16bf, au);
                __builtin_prefetch(wbase + ktStride, 0, 0);
#pragma unroll
                for (int t = 0; t < NTW; ++t) {
                    v16bf b = *(const v16bf*)(wbase + toff[t]);
                    acc[t] = __builtin_amdgcn_wmma_f32_16x16x32_bf16(
                        false, a, false, b, (short)0, acc[t], false, false);
                }
                wbase += ktStride;
            }
        }
    }

    // Epilogue. Lane covers column (l15) of each sub-tile; rows M = half*8 + r.
#pragma unroll
    for (int t = 0; t < NTW; ++t) {
        int nt = ntg * NTW + t;
        if (nt >= Ntiles) continue;
        int gn = nt * 16 + l15;
        if (gn >= Cout) continue;
        float bi = bias ? bias[gn] : 0.0f;
        bool  bn = (bn_g != nullptr);
        float mm = 0.f, bb = 0.f, sc = 0.f;
        if (bn) {
            mm = bn_m[gn]; bb = bn_b[gn];
            sc = bn_g[gn] * rsqrtf(bn_v[gn] + BN_EPS_F);
        }
#pragma unroll
        for (int r = 0; r < 8; ++r) {
            int   p = mtile * 16 + half * 8 + r;
            float x = acc[t][r] + bi;
            if (do_relu) x = fmaxf(x, 0.0f);
            if (bn)      x = (x - mm) * sc + bb;
            size_t oi = (size_t)p * outStride + chanOff + gn;
            if (out_f32)  out_f32[oi]  = x;
            if (out_bf16) out_bf16[oi] = f2bf(x);
        }
    }
}

// ---------------------------------------------------------------------------
// fused = a + bilinear_resize(small, Sin -> Sout), output bf16
// ---------------------------------------------------------------------------
__global__ void fuse_resize_add_kernel(const float* __restrict__ a,
                                       const float* __restrict__ small,
                                       int Sin, int Sout, int C,
                                       unsigned short* __restrict__ out) {
    int i = blockIdx.x * blockDim.x + threadIdx.x;
    int total = Sout * Sout * C;
    if (i >= total) return;
    int c = i % C;
    int x = (i / C) % Sout;
    int y = i / (C * Sout);
    float scale = (float)Sin / (float)Sout;
    float sy = fminf(fmaxf((y + 0.5f) * scale - 0.5f, 0.0f), (float)(Sin - 1));
    float sx = fminf(fmaxf((x + 0.5f) * scale - 0.5f, 0.0f), (float)(Sin - 1));
    int y0 = (int)sy, x0 = (int)sx;
    int y1 = min(y0 + 1, Sin - 1), x1 = min(x0 + 1, Sin - 1);
    float fy = sy - y0, fx = sx - x0;
    float v00 = small[((size_t)y0 * Sin + x0) * C + c];
    float v01 = small[((size_t)y0 * Sin + x1) * C + c];
    float v10 = small[((size_t)y1 * Sin + x0) * C + c];
    float v11 = small[((size_t)y1 * Sin + x1) * C + c];
    float top = v00 + (v01 - v00) * fx;
    float bot = v10 + (v11 - v10) * fx;
    out[i] = f2bf(a[i] + (top + (bot - top) * fy));
}

// ---------------------------------------------------------------------------
// Proposal scoring: softmax, anchor decode, clip, min-size filter.
// Emits boxes[NPROP][4] and 64-bit sort keys (score-desc, index-asc).
// ---------------------------------------------------------------------------
__global__ void proposals_kernel(const float* __restrict__ rpn_cls,
                                 const float* __restrict__ rpn_box,
                                 float* __restrict__ boxes,
                                 unsigned long long* __restrict__ keys) {
    int i = blockIdx.x * blockDim.x + threadIdx.x;
    if (i >= SORT_N) return;
    if (i >= NPROP) { keys[i] = 0xFFFFFFFFull; return; }  // pad: smallest, idx 0

    float c0 = rpn_cls[2 * i], c1 = rpn_cls[2 * i + 1];
    float mx = fmaxf(c0, c1);
    float e0 = __expf(c0 - mx), e1 = __expf(c1 - mx);
    float sc = e1 / (e0 + e1);

    int a  = i % 9;
    int hw = i / 9;
    int gx = hw % 64, gy = hw / 64;
    int ri = a / 3, si = a % 3;
    float r  = (ri == 0) ? 0.5f : ((ri == 1) ? 1.0f : 2.0f);
    float ws = roundf(sqrtf(64.0f / r));
    float hs = roundf(ws * r);
    float s  = 8.0f * (float)(1 << si);
    float W_ = ws * s, H_ = hs * s;
    float cxa = 3.5f + gx * 8.0f, cya = 3.5f + gy * 8.0f;
    float ax1 = cxa - 0.5f * (W_ - 1.0f), ay1 = cya - 0.5f * (H_ - 1.0f);
    float ax2 = cxa + 0.5f * (W_ - 1.0f), ay2 = cya + 0.5f * (H_ - 1.0f);

    float aw  = ax2 - ax1 + 1.0f, ah = ay2 - ay1 + 1.0f;
    float acx = ax1 + 0.5f * aw,  acy = ay1 + 0.5f * ah;
    float dx = rpn_box[4 * i],     dy = rpn_box[4 * i + 1];
    float dw = fminf(fmaxf(rpn_box[4 * i + 2], -BBOX_CLIP_F), BBOX_CLIP_F);
    float dh = fminf(fmaxf(rpn_box[4 * i + 3], -BBOX_CLIP_F), BBOX_CLIP_F);
    float pcx = dx * aw + acx, pcy = dy * ah + acy;
    float pw  = __expf(dw) * aw, ph = __expf(dh) * ah;
    float x1 = fminf(fmaxf(pcx - 0.5f * pw, 0.0f), (float)(IMG_WH - 1));
    float y1 = fminf(fmaxf(pcy - 0.5f * ph, 0.0f), (float)(IMG_WH - 1));
    float x2 = fminf(fmaxf(pcx + 0.5f * pw, 0.0f), (float)(IMG_WH - 1));
    float y2 = fminf(fmaxf(pcy + 0.5f * ph, 0.0f), (float)(IMG_WH - 1));
    if ((x2 - x1 + 1.0f) < MIN_SZ || (y2 - y1 + 1.0f) < MIN_SZ) sc = -1.0f;

    boxes[4 * i + 0] = x1; boxes[4 * i + 1] = y1;
    boxes[4 * i + 2] = x2; boxes[4 * i + 3] = y2;

    unsigned u = __float_as_uint(sc);
    u ^= (u >> 31) ? 0xFFFFFFFFu : 0x80000000u;   // monotone float->uint
    keys[i] = ((unsigned long long)u << 32) | (unsigned)(0xFFFFFFFFu - (unsigned)i);
}

// bitonic sort step, overall descending
__global__ void bitonic_kernel(unsigned long long* __restrict__ keys, int j, int k) {
    int i = blockIdx.x * blockDim.x + threadIdx.x;
    int ixj = i ^ j;
    if (ixj > i) {
        unsigned long long a = keys[i], b = keys[ixj];
        bool descSeg = ((i & k) == 0);
        bool swap = descSeg ? (a < b) : (a > b);
        if (swap) { keys[i] = b; keys[ixj] = a; }
    }
}

__global__ void gather_top_kernel(const unsigned long long* __restrict__ keys,
                                  const float* __restrict__ boxes,
                                  float* __restrict__ topb) {
    int i = blockIdx.x * blockDim.x + threadIdx.x;
    if (i >= PRE_NMS_N) return;
    unsigned idx = 0xFFFFFFFFu - (unsigned)(keys[i] & 0xFFFFFFFFull);
#pragma unroll
    for (int c = 0; c < 4; ++c) topb[4 * i + c] = boxes[4 * (size_t)idx + c];
}

// ---------------------------------------------------------------------------
// Single-workgroup NMS over 2000 boxes held in LDS, then stable compaction
// (kept first in order, then unkept) to 300 ROIs: (0, x1, y1, x2, y2).
// ---------------------------------------------------------------------------
__global__ void nms_kernel(const float* __restrict__ topb, float* __restrict__ rois) {
    __shared__ float bx[PRE_NMS_N][4];
    __shared__ unsigned char keep[PRE_NMS_N];
    int t = threadIdx.x, nt = blockDim.x;
    for (int i = t; i < PRE_NMS_N; i += nt) {
        bx[i][0] = topb[4 * i + 0]; bx[i][1] = topb[4 * i + 1];
        bx[i][2] = topb[4 * i + 2]; bx[i][3] = topb[4 * i + 3];
        keep[i] = 1;
    }
    __syncthreads();
    for (int i = 0; i < PRE_NMS_N; ++i) {
        if (keep[i]) {
            float x1 = bx[i][0], y1 = bx[i][1], x2 = bx[i][2], y2 = bx[i][3];
            float ar = (x2 - x1 + 1.0f) * (y2 - y1 + 1.0f);
            for (int j = i + 1 + t; j < PRE_NMS_N; j += nt) {
                if (!keep[j]) continue;
                float iw = fmaxf(fminf(x2, bx[j][2]) - fmaxf(x1, bx[j][0]) + 1.0f, 0.0f);
                float ih = fmaxf(fminf(y2, bx[j][3]) - fmaxf(y1, bx[j][1]) + 1.0f, 0.0f);
                float inter = iw * ih;
                float arj = (bx[j][2] - bx[j][0] + 1.0f) * (bx[j][3] - bx[j][1] + 1.0f);
                if (inter / (ar + arj - inter) > NMS_TH) keep[j] = 0;
            }
        }
        __syncthreads();
    }
    if (t == 0) {
        int r = 0;
        for (int i = 0; i < PRE_NMS_N && r < POST_NMS_N; ++i)
            if (keep[i]) {
                rois[5 * r + 0] = 0.0f;
                rois[5 * r + 1] = bx[i][0]; rois[5 * r + 2] = bx[i][1];
                rois[5 * r + 3] = bx[i][2]; rois[5 * r + 4] = bx[i][3];
                ++r;
            }
        for (int i = 0; i < PRE_NMS_N && r < POST_NMS_N; ++i)
            if (!keep[i]) {
                rois[5 * r + 0] = 0.0f;
                rois[5 * r + 1] = bx[i][0]; rois[5 * r + 2] = bx[i][1];
                rois[5 * r + 3] = bx[i][2]; rois[5 * r + 4] = bx[i][3];
                ++r;
            }
    }
}

// ---------------------------------------------------------------------------
// Host orchestration
// ---------------------------------------------------------------------------
static inline size_t alignup(size_t x) { return (x + 255) & ~(size_t)255; }

extern "C" void kernel_launch(void* const* d_in, const int* in_sizes, int n_in,
                              void* d_out, int out_size, void* d_ws, size_t ws_size,
                              hipStream_t stream) {
    (void)in_sizes; (void)n_in; (void)out_size; (void)ws_size;
    const float* fea3   = (const float*)d_in[0];
    const float* fea4   = (const float*)d_in[1];
    const float* fea5   = (const float*)d_in[2];
    const float* conv1w = (const float*)d_in[3];  const float* conv1b = (const float*)d_in[4];
    const float* bn1g = (const float*)d_in[5], *bn1b = (const float*)d_in[6];
    const float* bn1m = (const float*)d_in[7], *bn1v = (const float*)d_in[8];
    const float* conv2w = (const float*)d_in[9];  const float* conv2b = (const float*)d_in[10];
    const float* bn2g = (const float*)d_in[11], *bn2b = (const float*)d_in[12];
    const float* bn2m = (const float*)d_in[13], *bn2v = (const float*)d_in[14];
    const float* iw1 = (const float*)d_in[15], *ib1 = (const float*)d_in[16];
    const float* iw3 = (const float*)d_in[17], *ib3 = (const float*)d_in[18];
    const float* iw5 = (const float*)d_in[19], *ib5 = (const float*)d_in[20];
    const float* iwp = (const float*)d_in[21], *ibp = (const float*)d_in[22];
    const float* rpnw = (const float*)d_in[23], *rpnb = (const float*)d_in[24];
    const float* clsw = (const float*)d_in[25], *clsb = (const float*)d_in[26];
    const float* boxw = (const float*)d_in[27], *boxb = (const float*)d_in[28];
    const float* c3w = (const float*)d_in[29], *c3b = (const float*)d_in[30];
    const float* c4w = (const float*)d_in[31], *c4b = (const float*)d_in[32];

    float* out = (float*)d_out;
    // output layout (flat, in return order)
    const size_t PS_OFF  = 1500;                   // 300*5
    const size_t BB_OFF  = PS_OFF + 64 * 64 * 196;
    const size_t CLS_OFF = BB_OFF + 64 * 64 * 196;
    const size_t BOX_OFF = CLS_OFF + 64 * 64 * 18;

    // ---- workspace bump allocator ----
    char* p = (char*)d_ws;
    auto grab = [&](size_t bytes) { char* r = p; p += alignup(bytes); return r; };
    unsigned short* fea3_bf  = (unsigned short*)grab((size_t)64 * 64 * 512 * 2);
    unsigned short* fea5_bf  = (unsigned short*)grab((size_t)16 * 16 * 2048 * 2);
    float*          f3       = (float*)grab((size_t)64 * 64 * 1024 * 4);
    float*          f5       = (float*)grab((size_t)16 * 16 * 1024 * 4);
    unsigned short* fused1   = (unsigned short*)grab((size_t)64 * 64 * 1024 * 2);
    unsigned short* fused2   = (unsigned short*)grab((size_t)64 * 64 * 1024 * 2);
    unsigned short* concat   = (unsigned short*)grab((size_t)64 * 64 * 768 * 2);
    unsigned short* inc1     = (unsigned short*)grab((size_t)64 * 64 * 256 * 2);
    unsigned short* inc2     = (unsigned short*)grab((size_t)64 * 64 * 256 * 2);
    unsigned short* rpn_feat = (unsigned short*)grab((size_t)64 * 64 * 512 * 2);
    // packed weights: [Kt][Nt][32][16] bf16
    auto wbytes = [](int Kt, int Nt) { return (size_t)Kt * Nt * 512 * 2; };
    unsigned short* pw_c1  = (unsigned short*)grab(wbytes(16, 64));
    unsigned short* pw_c2  = (unsigned short*)grab(wbytes(64, 64));
    unsigned short* pw_i1  = (unsigned short*)grab(wbytes(32, 16));
    unsigned short* pw_i3  = (unsigned short*)grab(wbytes(288, 16));
    unsigned short* pw_i5  = (unsigned short*)grab(wbytes(800, 16));
    unsigned short* pw_ip  = (unsigned short*)grab(wbytes(24, 16));
    unsigned short* pw_rpn = (unsigned short*)grab(wbytes(72, 32));
    unsigned short* pw_cls = (unsigned short*)grab(wbytes(16, 2));
    unsigned short* pw_box = (unsigned short*)grab(wbytes(16, 3));
    unsigned short* pw_c3  = (unsigned short*)grab(wbytes(8, 13));
    unsigned short* pw_c4  = (unsigned short*)grab(wbytes(8, 13));
    float*              boxesb = (float*)grab((size_t)NPROP * 4 * 4);
    unsigned long long* keys   = (unsigned long long*)grab((size_t)SORT_N * 8);
    float*              topb   = (float*)grab((size_t)PRE_NMS_N * 4 * 4);

    // ---- activations to bf16 ----
    {
        int n3 = 64 * 64 * 512, n5 = 16 * 16 * 2048;
        cvt_bf16_kernel<<<(n3 + 255) / 256, 256, 0, stream>>>(fea3, fea3_bf, n3);
        cvt_bf16_kernel<<<(n5 + 255) / 256, 256, 0, stream>>>(fea5, fea5_bf, n5);
    }
    // ---- pack all weights ----
    auto pack = [&](const float* w, unsigned short* o, int Ktot, int Cout, int Kt, int Nt) {
        pack_w_kernel<<<dim3(Kt, Nt), 32, 0, stream>>>(w, o, Ktot, Cout, Nt);
    };
    pack(conv1w, pw_c1, 512, 1024, 16, 64);
    pack(conv2w, pw_c2, 2048, 1024, 64, 64);
    pack(iw1, pw_i1, 1024, 256, 32, 16);
    pack(iw3, pw_i3, 9 * 1024, 256, 288, 16);
    pack(iw5, pw_i5, 25 * 1024, 256, 800, 16);
    pack(iwp, pw_ip, 768, 256, 24, 16);
    pack(rpnw, pw_rpn, 9 * 256, 512, 72, 32);
    pack(clsw, pw_cls, 512, 18, 16, 2);
    pack(boxw, pw_box, 512, 36, 16, 3);
    pack(c3w, pw_c3, 256, 196, 8, 13);
    pack(c4w, pw_c4, 256, 196, 8, 13);

    // one wave -> 16 pixels x 128 channels (8 accumulators)
    auto conv = [&](const unsigned short* act, int Slog2, int Cin,
                    const unsigned short* wp, int Kt, int Nt, int KH, int KW,
                    const float* bias, const float* g, const float* b,
                    const float* m, const float* v, int relu,
                    float* of32, unsigned short* obf, int Cout, int stride, int off) {
        int S = 1 << Slog2;
        dim3 grid(S * S / 16, (Nt + 7) / 8);
        conv_wmma_kernel<8><<<grid, 32, 0, stream>>>(
            act, Slog2, Cin, wp, Kt, Nt, KH, KW, bias, g, b, m, v, relu,
            of32, obf, Cout, stride, off);
    };

    // ---- backbone fusion ----
    conv(fea3_bf, 6, 512, pw_c1, 16, 64, 1, 1, conv1b, bn1g, bn1b, bn1m, bn1v, 1,
         f3, nullptr, 1024, 1024, 0);
    conv(fea5_bf, 4, 2048, pw_c2, 64, 64, 1, 1, conv2b, bn2g, bn2b, bn2m, bn2v, 1,
         f5, nullptr, 1024, 1024, 0);
    {
        int n = 64 * 64 * 1024;
        fuse_resize_add_kernel<<<(n + 255) / 256, 256, 0, stream>>>(f3, fea4, 32, 64, 1024, fused1);
        fuse_resize_add_kernel<<<(n + 255) / 256, 256, 0, stream>>>(f3, f5, 16, 64, 1024, fused2);
    }

    // ---- inception (shared weights) on fused1 -> inc1, then fused2 -> inc2 ----
    for (int pass = 0; pass < 2; ++pass) {
        const unsigned short* fin = pass ? fused2 : fused1;
        unsigned short*       res = pass ? inc2 : inc1;
        conv(fin, 6, 1024, pw_i1, 32, 16, 1, 1, ib1, nullptr, nullptr, nullptr, nullptr, 1,
             nullptr, concat, 256, 768, 0);
        conv(fin, 6, 1024, pw_i3, 288, 16, 3, 3, ib3, nullptr, nullptr, nullptr, nullptr, 1,
             nullptr, concat, 256, 768, 256);
        conv(fin, 6, 1024, pw_i5, 800, 16, 5, 5, ib5, nullptr, nullptr, nullptr, nullptr, 1,
             nullptr, concat, 256, 768, 512);
        conv(concat, 6, 768, pw_ip, 24, 16, 1, 1, ibp, nullptr, nullptr, nullptr, nullptr, 1,
             nullptr, res, 256, 256, 0);
    }

    // ---- RPN head ----
    conv(inc1, 6, 256, pw_rpn, 72, 32, 3, 3, rpnb, nullptr, nullptr, nullptr, nullptr, 1,
         nullptr, rpn_feat, 512, 512, 0);
    conv(rpn_feat, 6, 512, pw_cls, 16, 2, 1, 1, clsb, nullptr, nullptr, nullptr, nullptr, 0,
         out + CLS_OFF, nullptr, 18, 18, 0);
    conv(rpn_feat, 6, 512, pw_box, 16, 3, 1, 1, boxb, nullptr, nullptr, nullptr, nullptr, 0,
         out + BOX_OFF, nullptr, 36, 36, 0);

    // ---- position-sensitive maps from inc2 ----
    conv(inc2, 6, 256, pw_c3, 8, 13, 1, 1, c3b, nullptr, nullptr, nullptr, nullptr, 0,
         out + PS_OFF, nullptr, 196, 196, 0);
    conv(inc2, 6, 256, pw_c4, 8, 13, 1, 1, c4b, nullptr, nullptr, nullptr, nullptr, 0,
         out + BB_OFF, nullptr, 196, 196, 0);

    // ---- proposals: score/decode -> sort -> top-2000 -> NMS -> 300 ROIs ----
    proposals_kernel<<<SORT_N / 256, 256, 0, stream>>>(out + CLS_OFF, out + BOX_OFF,
                                                       boxesb, keys);
    for (int k = 2; k <= SORT_N; k <<= 1)
        for (int j = k >> 1; j > 0; j >>= 1)
            bitonic_kernel<<<SORT_N / 256, 256, 0, stream>>>(keys, j, k);
    gather_top_kernel<<<(PRE_NMS_N + 255) / 256, 256, 0, stream>>>(keys, boxesb, topb);
    nms_kernel<<<1, 256, 0, stream>>>(topb, out);
}
